// MeanAggregator_5188320494244
// MI455X (gfx1250) — matrix operand order repaired and checked by reading, not compiled, not run on hardware
//
#include <hip/hip_runtime.h>

typedef __attribute__((ext_vector_type(16))) __bf16 v16bf;
typedef __attribute__((ext_vector_type(2)))  __bf16 v2bf;
typedef __attribute__((ext_vector_type(8)))  float  v8f;

#if defined(__has_builtin)
#  if __has_builtin(__builtin_amdgcn_cvt_pk_bf16_f32)
#    define HAVE_CVT_PK_BF16 1
#  endif
#endif
#ifndef HAVE_CVT_PK_BF16
#  define HAVE_CVT_PK_BF16 0
#endif

__device__ __forceinline__ unsigned pack2_bf16(float lo, float hi) {
#if HAVE_CVT_PK_BF16
  // single v_cvt_pk_bf16_f32: convert + pack in one VALU op
  v2bf p = __builtin_amdgcn_cvt_pk_bf16_f32(lo, hi);
  return __builtin_bit_cast(unsigned, p);
#else
  unsigned short l = __builtin_bit_cast(unsigned short, (__bf16)lo);
  unsigned short h = __builtin_bit_cast(unsigned short, (__bf16)hi);
  return (unsigned)l | ((unsigned)h << 16);
#endif
}

union AOp { unsigned u[8]; v16bf v; };
union BOp { uint4 q[2];   v16bf v; };

constexpr int BATCH   = 8;
constexpr int NN      = 4096;
constexpr int DD      = 64;
constexpr int KCHUNKS = NN / 32;   // 128
constexpr int CT      = DD / 16;   // 4 column tiles
constexpr int MT      = 2;         // 2 row tiles (32 rows) per wave

// ---------------------------------------------------------------------------
// Pass 1: repack x (f32, row-major N x D) into bf16 WMMA B-operand blocks.
// Block index = ((b*128 + kc)*4 + ct); block = 1 KB = 32 lanes x 8 dwords.
// Lane L covers n = ct*16 + (L&15), K range kc*32 + (L&16 ? 16 : 0) .. +15,
// dword r packs (K=2r, K=2r+1). Total: 4 MB in d_ws.
// ---------------------------------------------------------------------------
__global__ void prepack_x_kernel(const float* __restrict__ x,
                                 unsigned* __restrict__ xp) {
  int t    = blockIdx.x * blockDim.x + threadIdx.x;   // 131072 threads
  int lane = t & 31;
  int blk  = t >> 5;                                  // 0..4095
  int ct   = blk & 3;
  int kc   = (blk >> 2) & (KCHUNKS - 1);
  int b    = blk >> 9;
  int n     = ct * 16 + (lane & 15);
  int kbase = kc * 32 + ((lane & 16) ? 16 : 0);
  const float* xb = x + (size_t)b * NN * DD;
  unsigned* dst = xp + (size_t)blk * 256 + lane * 8;
#pragma unroll
  for (int r = 0; r < 8; ++r) {
    float lo = xb[(size_t)(kbase + 2 * r)     * DD + n];
    float hi = xb[(size_t)(kbase + 2 * r + 1) * DD + n];
    dst[r] = pack2_bf16(lo, hi);
  }
}

// ---------------------------------------------------------------------------
// Pass 2: fused  Y = (A @ x) / clamp(rowsum_f32(A), 1)
// One wave = 32 output rows x 64 cols, full K=4096 sweep.
// ---------------------------------------------------------------------------
__global__ void __launch_bounds__(256)
mean_agg_kernel(const float* __restrict__ A,
                const unsigned* __restrict__ xp,
                float* __restrict__ out) {
  const int lane  = threadIdx.x & 31;
  const int gwave = blockIdx.x * (blockDim.x >> 5) + (threadIdx.x >> 5);
  const int b     = gwave >> 7;       // 128 waves per batch
  const int mblk  = gwave & 127;
  const int mbase = mblk * (16 * MT); // 32 rows
  const int lrow  = lane & 15;
  const int khA   = (lane & 16) ? 16 : 0;  // A-operand lane K-offset

  const float* Ab  = A + (size_t)b * NN * NN;
  const uint4* xp4 = reinterpret_cast<const uint4*>(xp);

  const float* arow[MT];
#pragma unroll
  for (int mt = 0; mt < MT; ++mt)
    arow[mt] = Ab + (size_t)(mbase + 16 * mt + lrow) * NN;

  v8f   acc[MT][CT] = {};
  float rowsum[MT]  = {};

  const size_t bblk_base = (size_t)b * KCHUNKS * CT;

  for (int k0 = 0; k0 < NN; k0 += 32) {
    // ---- B operands: 2 contiguous b128 loads each (prepacked, L2-hot) ----
    BOp bop[CT];
    const size_t bblk = bblk_base + (size_t)(k0 >> 5) * CT;
#pragma unroll
    for (int ct = 0; ct < CT; ++ct) {
      const uint4* bp = xp4 + (bblk + ct) * 64 + lane * 2;
      bop[ct].q[0] = bp[0];
      bop[ct].q[1] = bp[1];
    }

#pragma unroll
    for (int mt = 0; mt < MT; ++mt) {
      // ---- A tile: 16 f32 per lane in operand order, 4x b128 ----
      const float4* ap = reinterpret_cast<const float4*>(arow[mt] + k0 + khA);
      float4 f0 = ap[0], f1 = ap[1];   // K = k0+khA   + 0..7
      float4 f2 = ap[2], f3 = ap[3];   // K = k0+khA   + 8..15

      // f32 row-sum accumulation (exact deg), co-executes with WMMA
      rowsum[mt] += (f0.x + f0.y + f0.z + f0.w) + (f1.x + f1.y + f1.z + f1.w)
                  + (f2.x + f2.y + f2.z + f2.w) + (f3.x + f3.y + f3.z + f3.w);

      AOp a;
      a.u[0] = pack2_bf16(f0.x, f0.y);  a.u[1] = pack2_bf16(f0.z, f0.w);
      a.u[2] = pack2_bf16(f1.x, f1.y);  a.u[3] = pack2_bf16(f1.z, f1.w);
      a.u[4] = pack2_bf16(f2.x, f2.y);  a.u[5] = pack2_bf16(f2.z, f2.w);
      a.u[6] = pack2_bf16(f3.x, f3.y);  a.u[7] = pack2_bf16(f3.z, f3.w);

#pragma unroll
      for (int ct = 0; ct < CT; ++ct)
        acc[mt][ct] = __builtin_amdgcn_wmma_f32_16x16x32_bf16(
            false, a.v, false, bop[ct].v, (short)0, acc[mt][ct], false, false);

      // prefetch the A stream ahead (speculative; dropped if OOB)
      __builtin_prefetch(arow[mt] + k0 + 256 + khA, 0, 0);
    }
  }

  // ---- epilogue: deg = clamp(rowsum, 1); divide and store ----
#pragma unroll
  for (int mt = 0; mt < MT; ++mt) {
    // lanes L and L^16 hold the two K-halves of row (L&15)
    float s = rowsum[mt] + __shfl_xor(rowsum[mt], 16, 32);
#pragma unroll
    for (int r = 0; r < 8; ++r) {
      const int rowInTile = r + ((lane & 16) ? 8 : 0);  // C/D layout M index
      float d   = __shfl(s, rowInTile, 32);
      float inv = 1.0f / fmaxf(d, 1.0f);
      const int row = mbase + 16 * mt + rowInTile;
      float* orow = out + ((size_t)b * NN + row) * DD;
#pragma unroll
      for (int ct = 0; ct < CT; ++ct)
        orow[ct * 16 + (lane & 15)] = acc[mt][ct][r] * inv;
    }
  }
}

// ---------------------------------------------------------------------------
// Host launcher. d_in[0]=x (8*4096*64 f32), d_in[1]=A (8*4096*4096 f32).
// d_ws needs 4 MB for the bf16-prepacked x.
// ---------------------------------------------------------------------------
extern "C" void kernel_launch(void* const* d_in, const int* in_sizes, int n_in,
                              void* d_out, int out_size, void* d_ws, size_t ws_size,
                              hipStream_t stream) {
  const float* x = (const float*)d_in[0];
  const float* A = (const float*)d_in[1];
  float* out     = (float*)d_out;
  unsigned* xp   = (unsigned*)d_ws;   // 4 MB used

  // Pass 1: 4096 blocks' worth of packing = 131072 threads
  prepack_x_kernel<<<512, 256, 0, stream>>>(x, xp);

  // Pass 2: 1024 waves = 128 blocks x 8 waves
  mean_agg_kernel<<<128, 256, 0, stream>>>(A, xp, out);
}
